// QuantumHybrid_65481071395430
// MI455X (gfx1250) — matrix operand order repaired and checked by reading, not compiled, not run on hardware
//
#include <hip/hip_runtime.h>
#include <hip/hip_bf16.h>
#include <stdint.h>

#define N_WIRES    10
#define N_AMPS     1024
#define N_RAND_OPS 30
#define SIM_THREADS 512
#define N_WAVES    (SIM_THREADS / 32)
#define FEAT_STRIDE 16
#define SHIFT_F 1.57079632679489662f

typedef __attribute__((ext_vector_type(16))) _Float16 v16h;
typedef __attribute__((ext_vector_type(8)))  float    v8f;

struct GateSpec { int g[N_RAND_OPS]; int w0[N_RAND_OPS]; int w1[N_RAND_OPS]; };

// ---------------- device: statevector gates in LDS ----------------
// Wire w corresponds to bit (N_WIRES-1-w) of the amplitude index
// (axis 1 of the reshaped state is the MSB).

__device__ __forceinline__ void pair_idx(int t, int w, unsigned& i0, unsigned& i1) {
  unsigned mask = 1u << (N_WIRES - 1 - w);
  unsigned lo = mask - 1u;
  i0 = ((unsigned)t & lo) | (((unsigned)t & ~lo) << 1);
  i1 = i0 | mask;
}

__device__ __forceinline__ void gate_ry(float* re, float* im, int t, int w, float c, float s) {
  unsigned i0, i1; pair_idx(t, w, i0, i1);
  float r0 = re[i0], m0 = im[i0], r1 = re[i1], m1 = im[i1];
  re[i0] = c * r0 - s * r1;  im[i0] = c * m0 - s * m1;
  re[i1] = s * r0 + c * r1;  im[i1] = s * m0 + c * m1;
}

__device__ __forceinline__ void gate_rx(float* re, float* im, int t, int w, float c, float s) {
  unsigned i0, i1; pair_idx(t, w, i0, i1);
  float r0 = re[i0], m0 = im[i0], r1 = re[i1], m1 = im[i1];
  re[i0] = c * r0 + s * m1;  im[i0] = c * m0 - s * r1;   // a0' = c*a0 - i*s*a1
  re[i1] = c * r1 + s * m0;  im[i1] = c * m1 - s * r0;   // a1' = -i*s*a0 + c*a1
}

__device__ __forceinline__ void gate_rz(float* re, float* im, int t, int w, float c, float s) {
  unsigned i0, i1; pair_idx(t, w, i0, i1);
  float r0 = re[i0], m0 = im[i0], r1 = re[i1], m1 = im[i1];
  re[i0] = c * r0 + s * m0;  im[i0] = c * m0 - s * r0;   // * e^{-i t/2}
  re[i1] = c * r1 - s * m1;  im[i1] = c * m1 + s * r1;   // * e^{+i t/2}
}

__device__ __forceinline__ void gate_cnot(float* re, float* im, int t, int wc, int wt) {
  unsigned i0, i1; pair_idx(t, wt, i0, i1);
  unsigned cm = 1u << (N_WIRES - 1 - wc);
  if (i0 & cm) {
    float tr = re[i0]; re[i0] = re[i1]; re[i1] = tr;
    float ti = im[i0]; im[i0] = im[i1]; im[i1] = ti;
  }
}

// General 2x2 complex gate: [a b; c d] (used for fused RY(ty)*RX(tx))
__device__ __forceinline__ void gate_u2(float* re, float* im, int t, int w,
                                        float ar, float ai, float br, float bi,
                                        float cr, float ci, float dr, float di) {
  unsigned i0, i1; pair_idx(t, w, i0, i1);
  float r0 = re[i0], m0 = im[i0], r1 = re[i1], m1 = im[i1];
  re[i0] = ar * r0 - ai * m0 + br * r1 - bi * m1;
  im[i0] = ar * m0 + ai * r0 + br * m1 + bi * r1;
  re[i1] = cr * r0 - ci * m0 + dr * r1 - di * m1;
  im[i1] = cr * m0 + ci * r0 + dr * m1 + di * r1;
}

__global__ __launch_bounds__(SIM_THREADS)
void qsim_kernel(const float* __restrict__ xb, const float* __restrict__ rx_t,
                 const float* __restrict__ ry_t, const float* __restrict__ rp,
                 float* __restrict__ feats, GateSpec spec) {
  __shared__ float re[N_AMPS];
  __shared__ float im[N_AMPS];
  __shared__ float part[N_WAVES][N_WIRES];
  const int t = threadIdx.x;
  const int b = blockIdx.x;

  // |0...0>
  re[t] = 0.f; im[t] = 0.f; re[t + 512] = 0.f; im[t + 512] = 0.f;
  if (t == 0) re[0] = 1.f;
  __syncthreads();

  // Encoder: RY(x[b,w]) on each wire (per-sample angles)
  #pragma unroll
  for (int w = 0; w < N_WIRES; ++w) {
    float s, c; __sincosf(0.5f * xb[b * N_WIRES + w], &s, &c);
    gate_ry(re, im, t, w, c, s);
    __syncthreads();
  }

  // Fixed random layer (spec replicated from numpy RandomState(42))
  for (int k = 0; k < N_RAND_OPS; ++k) {
    int g = spec.g[k];
    if (g == 3) {
      gate_cnot(re, im, t, spec.w0[k], spec.w1[k]);
    } else {
      float s, c; __sincosf(0.5f * rp[k], &s, &c);
      if (g == 0)      gate_rx(re, im, t, spec.w0[k], c, s);
      else if (g == 1) gate_ry(re, im, t, spec.w0[k], c, s);
      else             gate_rz(re, im, t, spec.w0[k], c, s);
    }
    __syncthreads();
  }

  // Shared trainable RX then RY per wire, fused into one pass: U = RY(ty)*RX(tx)
  // U00 = cy*cx + i*sy*sx   U01 = -sy*cx - i*cy*sx
  // U10 = sy*cx - i*cy*sx   U11 =  cy*cx - i*sy*sx
  {
    float sx, cx, sy, cy;
    __sincosf(0.5f * rx_t[0], &sx, &cx);
    __sincosf(0.5f * ry_t[0], &sy, &cy);
    float ar =  cy * cx, ai =  sy * sx;
    float br = -sy * cx, bi = -cy * sx;
    float cr =  sy * cx, ci = -cy * sx;
    float dr =  cy * cx, di = -sy * sx;
    #pragma unroll
    for (int w = 0; w < N_WIRES; ++w) {
      gate_u2(re, im, t, w, ar, ai, br, bi, cr, ci, dr, di);
      __syncthreads();
    }
  }

  // MeasureAll PauliZ: feats[w] = sum_i p_i * (bit_w(i) ? -1 : +1)
  // Per-thread partials for all 10 wires -> in-wave shuffle butterfly -> LDS.
  float p0 = re[t] * re[t] + im[t] * im[t];
  float p1 = re[t + 512] * re[t + 512] + im[t + 512] * im[t + 512];
  float acc[N_WIRES];
  #pragma unroll
  for (int w = 0; w < N_WIRES; ++w) {
    unsigned mask = 1u << (N_WIRES - 1 - w);
    acc[w] = (((unsigned)t & mask) ? -p0 : p0)
           + (((unsigned)(t + 512) & mask) ? -p1 : p1);
  }
  #pragma unroll
  for (int off = 16; off > 0; off >>= 1) {
    #pragma unroll
    for (int w = 0; w < N_WIRES; ++w)
      acc[w] += __shfl_xor(acc[w], off, 32);
  }
  const int wv = t >> 5;
  const int lane = t & 31;
  if (lane == 0) {
    #pragma unroll
    for (int w = 0; w < N_WIRES; ++w) part[wv][w] = acc[w];
  }
  __syncthreads();
  if (t < N_WIRES) {
    float s = 0.f;
    #pragma unroll
    for (int v = 0; v < N_WAVES; ++v) s += part[v][t];
    feats[b * FEAT_STRIDE + t] = s;
  }
}

// ---------------- device: WMMA head (GEMV via 16x16x32 f16) ----------------
// One wave computes 16 rows of logits: D = A(16x32) * B(32x16), column N=0
// holds feats . head_w (K padded with zeros past 10).

__global__ __launch_bounds__(256)
void head_kernel(const float* __restrict__ feats, const float* __restrict__ hw,
                 const float* __restrict__ hb, float* __restrict__ out, int bsz) {
  int wave = (int)((blockIdx.x * blockDim.x + threadIdx.x) >> 5);
  int lane = (int)(threadIdx.x & 31);
  int row0 = wave * 16;
  if (row0 >= bsz) return;            // wave-uniform: EXEC stays all-ones for WMMA
  int half = lane >> 4;
  int m = lane & 15;
  float wb = hb[0];

  v16h a, bmat;
  #pragma unroll
  for (int j = 0; j < 16; ++j) {
    // A operand, contiguous k order chosen here (not the ISA table order): lanes0-15 K=0..15; lanes16-31 K=16..31
    int ka = j + (half ? 16 : 0);
    int row = row0 + m;
    float av = (ka < N_WIRES && row < bsz) ? feats[row * FEAT_STRIDE + ka] : 0.f;
    // B operand, same contiguous k order as A: lane n holds column N=n%16; lanes0-15 K=0..15, lanes16-31 K=16..31
    int kb = j + (half ? 16 : 0);
    float bv = (m == 0 && kb < N_WIRES) ? hw[kb] : 0.f;
    a[j]    = (_Float16)av;
    bmat[j] = (_Float16)bv;
  }

  v8f c = {};
  c = __builtin_amdgcn_wmma_f32_16x16x32_f16(false, a, false, bmat,
                                             (short)0, c, false, false);

  // C/D: lane0 -> (rows 0..7, N=0), lane16 -> (rows 8..15, N=0)
  if (m == 0) {
    #pragma unroll
    for (int r = 0; r < 8; ++r) {
      int row = row0 + half * 8 + r;
      if (row < bsz) {
        float logit = c[r] + wb + SHIFT_F;
        // sigmoid with fast hardware reciprocal (v_rcp_f32), ~1 ulp: plenty here
        out[row] = __builtin_amdgcn_rcpf(1.f + __expf(-logit));
      }
    }
  }
}

// ---------------- host: numpy RandomState(42) replica ----------------
namespace nprng {
struct MT19937 {
  uint32_t mt[624]; int idx;
  explicit MT19937(uint32_t s) {
    mt[0] = s;
    for (int i = 1; i < 624; ++i)
      mt[i] = 1812433253u * (mt[i - 1] ^ (mt[i - 1] >> 30)) + (uint32_t)i;
    idx = 624;
  }
  uint32_t next32() {
    if (idx >= 624) {
      for (int i = 0; i < 624; ++i) {
        uint32_t y = (mt[i] & 0x80000000u) | (mt[(i + 1) % 624] & 0x7fffffffu);
        uint32_t v = mt[(i + 397) % 624] ^ (y >> 1);
        mt[i] = (y & 1u) ? (v ^ 0x9908b0dfu) : v;
      }
      idx = 0;
    }
    uint32_t y = mt[idx++];
    y ^= y >> 11; y ^= (y << 7) & 0x9d2c5680u; y ^= (y << 15) & 0xefc60000u; y ^= y >> 18;
    return y;
  }
  uint64_t next64() { uint64_t hi = next32(); uint64_t lo = next32(); return (hi << 32) | lo; }
  // legacy RandomState.randint: masked rejection on 64-bit draws
  uint64_t randint(uint64_t n) {
    uint64_t rng = n - 1, mask = rng;
    mask |= mask >> 1; mask |= mask >> 2; mask |= mask >> 4;
    mask |= mask >> 8; mask |= mask >> 16; mask |= mask >> 32;
    uint64_t v; do { v = next64() & mask; } while (v > rng);
    return v;
  }
  // numpy random_interval: 32-bit draw path for max <= 0xffffffff (used by shuffle)
  uint64_t interval(uint64_t mx) {
    if (!mx) return 0;
    uint64_t mask = mx;
    mask |= mask >> 1; mask |= mask >> 2; mask |= mask >> 4;
    mask |= mask >> 8; mask |= mask >> 16; mask |= mask >> 32;
    uint64_t v;
    if (mx <= 0xffffffffULL) { do { v = next32() & mask; } while (v > mx); }
    else                     { do { v = next64() & mask; } while (v > mx); }
    return v;
  }
  void permutation10(int* arr) {   // choice(10, 2, replace=False) = permutation(10)[:2]
    for (int i = 0; i < 10; ++i) arr[i] = i;
    for (int i = 9; i >= 1; --i) {
      int j = (int)interval((uint64_t)i);
      int tmp = arr[i]; arr[i] = arr[j]; arr[j] = tmp;
    }
  }
};
} // namespace nprng

static void build_gate_spec(GateSpec& sp) {
  nprng::MT19937 rs(42u);
  for (int k = 0; k < N_RAND_OPS; ++k) {
    int g = (int)rs.randint(4);        // 0=rx 1=ry 2=rz 3=cnot
    sp.g[k] = g;
    if (g == 3) {
      int arr[10]; rs.permutation10(arr);
      sp.w0[k] = arr[0]; sp.w1[k] = arr[1];
    } else {
      sp.w0[k] = (int)rs.randint(10); sp.w1[k] = -1;
    }
  }
}

// ---------------- entry ----------------
extern "C" void kernel_launch(void* const* d_in, const int* in_sizes, int n_in,
                              void* d_out, int out_size, void* d_ws, size_t ws_size,
                              hipStream_t stream) {
  (void)n_in; (void)out_size; (void)ws_size;
  const float* xb  = (const float*)d_in[0];  // state_batch (bsz, 10)
  const float* rxt = (const float*)d_in[1];  // rx_theta (1,)
  const float* ryt = (const float*)d_in[2];  // ry_theta (1,)
  const float* rp  = (const float*)d_in[3];  // rand_params (30,)
  const float* hw  = (const float*)d_in[4];  // head_w (1, 10)
  const float* hb  = (const float*)d_in[5];  // head_b (1,)
  float* out   = (float*)d_out;              // (bsz,) float32
  float* feats = (float*)d_ws;               // bsz x FEAT_STRIDE floats scratch

  int bsz = in_sizes[0] / N_WIRES;

  GateSpec spec;
  build_gate_spec(spec);                     // deterministic each call

  qsim_kernel<<<bsz, SIM_THREADS, 0, stream>>>(xb, rxt, ryt, rp, feats, spec);

  int tiles   = (bsz + 15) / 16;             // one wave per 16-row tile
  int threads = 256;                         // 8 waves per block
  int blocks  = (tiles * 32 + threads - 1) / threads;
  head_kernel<<<blocks, threads, 0, stream>>>(feats, hw, hb, out, bsz);
}
